// PSKT_1194000908963
// MI455X (gfx1250) — matrix operand order repaired and checked by reading, not compiled, not run on hardware
//
#include <hip/hip_runtime.h>
#include <hip/hip_bf16.h>
#include <stdint.h>

// ============================================================================
// Knowledge-tracing RNN for MI455X (gfx1250).
//  - prep kernels: embedding gathers + f32->bf16 weight transpose ([N][K]).
//  - recur kernel: 16 persistent workgroups (one N=32 column slice each),
//    199 sequential scan steps + final prediction step. All GEMMs via
//    v_wmma_f32_16x16x32_bf16 (software-pipelined fragment loads); h staged
//    to LDS with async global->LDS loads; inter-WG sync = atomic counter +
//    s_cluster_barrier (NOP if unclustered).
//  Requires ws_size >= ~30 MB.
// ============================================================================

typedef unsigned short u16t;
typedef __attribute__((ext_vector_type(8)))  __bf16 v8bf;
typedef __attribute__((ext_vector_type(16))) __bf16 v16bf;
typedef __attribute__((ext_vector_type(8)))  float  v8f;
typedef int b128v __attribute__((vector_size(16)));   // matches async builtin param

#ifndef __has_builtin
#define __has_builtin(x) 0
#endif
#if __has_builtin(__builtin_amdgcn_global_load_async_to_lds_b128) && \
    __has_builtin(__builtin_amdgcn_s_wait_asynccnt)
#define HAVE_ASYNC 1
#else
#define HAVE_ASYNC 0
#endif

// ---------------- workspace layout (u16 elements) ----------------
static constexpr size_t O_Wskq = 0;
static constexpr size_t O_Wekq = O_Wskq + (size_t)512 * 768;
static constexpr size_t O_Wks1 = O_Wekq + (size_t)512 * 768;
static constexpr size_t O_Wks2 = O_Wks1 + (size_t)512 * 1024;
static constexpr size_t O_Wka2 = O_Wks2 + (size_t)512 * 1536;
static constexpr size_t O_Wki1 = O_Wka2 + (size_t)512 * 1024;
static constexpr size_t O_Wki2 = O_Wki1 + (size_t)512 * 1536;
static constexpr size_t O_Wfo  = O_Wki2 + (size_t)512 * 1280;
static constexpr size_t O_Wm1  = O_Wfo  + (size_t)512 * 1024;
static constexpr size_t O_qc   = O_Wm1  + (size_t)128 * 768;
static constexpr size_t O_qa   = O_qc  + (size_t)64 * 200 * 256;
static constexpr size_t O_tde  = O_qa  + (size_t)64 * 200 * 256;
static constexpr size_t O_skq  = O_tde + (size_t)64 * 199 * 256;
static constexpr size_t O_ekq  = O_skq  + (size_t)64 * 512;
static constexpr size_t O_resk = O_ekq  + (size_t)64 * 512;
static constexpr size_t O_ks   = O_resk + (size_t)64 * 512;
static constexpr size_t O_u    = O_ks   + (size_t)64 * 512;
static constexpr size_t O_ka   = O_u    + (size_t)64 * 512;
static constexpr size_t O_hbf  = O_ka   + (size_t)64 * 512;
static constexpr size_t O_m1a  = O_hbf  + (size_t)64 * 512;
static constexpr size_t U16_TOT = O_m1a + (size_t)64 * 128;
static constexpr size_t FBYTES  = U16_TOT * 2;           // float region after u16 region
static constexpr size_t F_hf    = 0;                     // h master (f32) [64][512]
static constexpr size_t F_diff  = F_hf + (size_t)64 * 512;
static constexpr size_t F_alpha = F_diff + (size_t)64 * 200;
static constexpr size_t F_TOT   = F_alpha + (size_t)64 * 200;
static constexpr int QCS = 200 * 256;   // row stride of qc/qa tables
static constexpr int TDS = 199 * 256;   // row stride of tde table

// ---------------- helpers ----------------
__device__ __forceinline__ u16t f2bf(float f) {
  unsigned u = __float_as_uint(f);
  unsigned r = (u + 0x7FFFu + ((u >> 16) & 1u)) >> 16;
  return (u16t)r;
}
__device__ __forceinline__ float bf2f(u16t h) {
  return __uint_as_float(((unsigned)h) << 16);
}
__device__ __forceinline__ float sigf(float x) { return 1.0f / (1.0f + __expf(-x)); }

__device__ __forceinline__ void lds_copy16(const u16t* g, u16t* l) {
#if HAVE_ASYNC
  __builtin_amdgcn_global_load_async_to_lds_b128(
      (__attribute__((address_space(1))) b128v*)(u16t*)g,
      (__attribute__((address_space(3))) b128v*)l, 0, 0);
#else
  *(float4*)(l) = *(const float4*)(g);
#endif
}

// D = A*B + D over a concatenated-A input of up to 3 pieces, software-
// pipelined by one k-step so fragment loads overlap the previous WMMA.
// A fragment layout (16-bit, 16x32): row = m-tile + (lane&15), chunks at
// k0 + (lane<16?0:8) and +16.  B fragment (32x16): col = n-tile + (lane&15),
// same K chunks as A: k0 + (lane<16?0:8) and +16.  Weights stored [N][K].
__device__ __forceinline__ v8f gemm_cat3(
    v8f acc, int Kdim,
    const u16t* p0, int s0, int l0,
    const u16t* p1, int s1, int l1,
    const u16t* p2, int s2,
    const u16t* Wt, int row, int ncol, int off0, int koffB) {
  const u16t* wb = Wt + (size_t)ncol * Kdim;
  auto loadf = [&](int k0, v16bf& av, v16bf& bv) {
    const u16t* ra;
    if (k0 < l0)      ra = p0 + (size_t)row * s0 + k0;
    else if (k0 < l1) ra = p1 + (size_t)row * s1 + (k0 - l0);
    else              ra = p2 + (size_t)row * s2 + (k0 - l1);
    v8bf lo = *(const v8bf*)(ra + off0);
    v8bf hi = *(const v8bf*)(ra + off0 + 16);
    av = __builtin_shufflevector(lo, hi, 0,1,2,3,4,5,6,7,8,9,10,11,12,13,14,15);
    v8bf blo = *(const v8bf*)(wb + k0 + off0);
    v8bf bhi = *(const v8bf*)(wb + k0 + off0 + 16);
    bv = __builtin_shufflevector(blo, bhi, 0,1,2,3,4,5,6,7,8,9,10,11,12,13,14,15);
  };
  v16bf a0, b0;
  loadf(0, a0, b0);
  for (int k0 = 32; k0 < Kdim; k0 += 32) {
    v16bf a1, b1;
    loadf(k0, a1, b1);                      // in flight during the WMMA below
    acc = __builtin_amdgcn_wmma_f32_16x16x32_bf16(false, a0, false, b0,
                                                  (short)0, acc, false, false);
    a0 = a1; b0 = b1;
  }
  acc = __builtin_amdgcn_wmma_f32_16x16x32_bf16(false, a0, false, b0,
                                                (short)0, acc, false, false);
  return acc;
}

// grid-wide sense barrier across the 16 persistent WGs (+cluster barrier,
// which is the fast path if launched as a cluster, NOP otherwise).
__device__ __forceinline__ void gbar(int* cnt, int target) {
  __threadfence();
  __syncthreads();
  if (threadIdx.x == 0) {
    atomicAdd(cnt, 1);
    while (__hip_atomic_load(cnt, __ATOMIC_RELAXED, __HIP_MEMORY_SCOPE_AGENT) < target)
      __builtin_amdgcn_s_sleep(2);
  }
  __syncthreads();
  __threadfence();
  __builtin_amdgcn_s_cluster_barrier();
  __syncthreads();
}

// ---------------- prep kernels ----------------
__global__ void kt_zero_cnt(int* c) { if (threadIdx.x == 0) *c = 0; }

// transpose+convert f32 [din][dout] -> bf16 [dout][din]
__global__ void kt_wprep(const float* __restrict__ src, u16t* __restrict__ dst,
                         int din, int dout) {
  int i = blockIdx.x * blockDim.x + threadIdx.x;
  if (i >= din * dout) return;
  int k = i / dout, n = i % dout;
  dst[(size_t)n * din + k] = f2bf(src[i]);
}

// per (b,t): qc, qa (=Er[R]), tde (=Etd[TD]), diff, alpha
__global__ void kt_prep_acts(
    const int* __restrict__ Q, const int* __restrict__ KC,
    const int* __restrict__ R, const int* __restrict__ TS,
    const float* __restrict__ Eq, const float* __restrict__ Ec,
    const float* __restrict__ Er, const float* __restrict__ Etd,
    const float* __restrict__ wdiff, const float* __restrict__ bdiff,
    const float* __restrict__ walpha, const float* __restrict__ balpha,
    const float* __restrict__ pa, const float* __restrict__ pb,
    u16t* __restrict__ qc_bf, u16t* __restrict__ qa_bf, u16t* __restrict__ tde_bf,
    float* __restrict__ diff_o, float* __restrict__ alpha_o) {
  int t = blockIdx.x, b = blockIdx.y, k = threadIdx.x;   // 200 x 64 blocks, 256 thr
  int q  = Q[b * 200 + t];
  int kc = KC[b * 200 + t];
  int r  = R[b * 200 + t];
  float qe = Eq[(size_t)q * 256 + k];
  float p1 = qe * wdiff[k];
  float p2 = qe * walpha[k];
  for (int o = 16; o > 0; o >>= 1) { p1 += __shfl_down(p1, o, 32); p2 += __shfl_down(p2, o, 32); }
  __shared__ float r1[8], r2[8];
  int wv = k >> 5, ln = k & 31;
  if (ln == 0) { r1[wv] = p1; r2[wv] = p2; }
  __syncthreads();
  if (k == 0) {
    float s1 = 0.f, s2 = 0.f;
    for (int i = 0; i < 8; ++i) { s1 += r1[i]; s2 += r2[i]; }
    r1[0] = sigf(s1 + bdiff[0]);
    r2[0] = sigf(s2 + balpha[0]);
  }
  __syncthreads();
  float diff = r1[0], alpha = r2[0];
  if (k == 0) { diff_o[b * 200 + t] = diff; alpha_o[b * 200 + t] = alpha; }
  float qcv = Ec[(size_t)kc * 256 + k] + qe * (pa[0] * diff + pb[0] * alpha);
  qc_bf[(size_t)b * QCS + (size_t)t * 256 + k] = f2bf(qcv);
  qa_bf[(size_t)b * QCS + (size_t)t * 256 + k] = f2bf(Er[(size_t)r * 256 + k]);
  if (t < 199) {
    int td = (TS[b * 200 + t + 1] - TS[b * 200 + t]) / 60;
    td = td < 0 ? 0 : (td > 43200 ? 43200 : td);
    tde_bf[(size_t)b * TDS + (size_t)t * 256 + k] = f2bf(Etd[(size_t)td * 256 + k]);
  }
}

// ---------------- persistent recurrent kernel ----------------
__global__ void __launch_bounds__(512, 1) kt_recur(
    u16t* __restrict__ U, float* __restrict__ F, int* cnt,
    const float* __restrict__ kinit,
    const float* __restrict__ bskq, const float* __restrict__ bekq,
    const float* __restrict__ bks1, const float* __restrict__ bks2,
    const float* __restrict__ bka2, const float* __restrict__ bki1,
    const float* __restrict__ bki2, const float* __restrict__ bfo,
    const float* __restrict__ bm1,
    const float* __restrict__ w2, const float* __restrict__ b2p,
    const float* __restrict__ Gw, const float* __restrict__ Gb,
    const float* __restrict__ Sw, const float* __restrict__ Sb,
    const int* __restrict__ KCp, float* __restrict__ out) {
  extern __shared__ __align__(16) u16t h_lds[];    // 64x512 bf16 = 64KB
  const int wg   = blockIdx.x;                     // 0..15, owns cols [wg*32, wg*32+32)
  const int wave = threadIdx.x >> 5;
  const int lane = threadIdx.x & 31;
  const int n0   = wg * 32;
  const int off0  = (lane < 16) ? 0 : 8;
  const int koffB = (lane < 16) ? 0 : 16;
  int target = 0;

  // init h = kinit broadcast over batch rows
  for (int i = wg * 512 + threadIdx.x; i < 64 * 512; i += 16 * 512) {
    float v = kinit[i & 511];
    F[F_hf + i] = v;
    U[O_hbf + i] = f2bf(v);
  }
  target += 16; gbar(cnt, target);

  for (int t = 0; t < 200; ++t) {
    // ---- stage 0: stage old h into LDS (async global->LDS) ----
    {
      const u16t* hsrc = U + O_hbf;
      for (int i = threadIdx.x; i < 4096; i += 512)
        lds_copy16(hsrc + (size_t)i * 8, h_lds + (size_t)i * 8);
#if HAVE_ASYNC
      __builtin_amdgcn_s_wait_asynccnt(0);
#endif
      __syncthreads();
    }
    const u16t* qcb = U + O_qc + (size_t)t * 256;
    // common per-wave tile mapping (8 tiles: 4 m-tiles x 2 n-tiles)
    const int tw = wave & 7;
    const int m0w = (tw >> 1) * 16;
    const int roww = m0w + (lane & 15);
    const int ncolw = n0 + (tw & 1) * 16 + (lane & 15);
    const int rbasew = m0w + ((lane < 16) ? 0 : 8);

    // ---- stage A: skq (waves 0-7) & ekq (waves 8-15); A = [qc_t | h] ----
    {
      const int gsel = wave >> 3;
      const u16t* Wt = U + (gsel ? O_Wekq : O_Wskq);
      const float* bias = gsel ? bekq : bskq;
      v8f acc = {};
      acc = gemm_cat3(acc, 768, qcb, QCS, 256, h_lds, 512, 768, h_lds, 512,
                      Wt, roww, ncolw, off0, koffB);
      u16t* dst = U + (gsel ? O_ekq : O_skq);
      float bb = bias[ncolw];
#pragma unroll
      for (int r = 0; r < 8; ++r)
        dst[(size_t)(rbasew + r) * 512 + ncolw] = f2bf(acc[r] + bb);
    }
    __threadfence();
    __syncthreads();
    // resk = ekq - skq on this WG's own columns
    for (int i = threadIdx.x; i < 2048; i += 512) {
      int rr = i >> 5, c = n0 + (i & 31);
      float e = bf2f(U[O_ekq + (size_t)rr * 512 + c]);
      float s = bf2f(U[O_skq + (size_t)rr * 512 + c]);
      U[O_resk + (size_t)rr * 512 + c] = f2bf(e - s);
    }
    target += 16; gbar(cnt, target);

    // ---- stage B: ks (waves 0-7), u=(tanh(ki1)+1)/2 (waves 8-15) ----
    if (wave < 8) {
      v8f a2 = {};
      a2 = gemm_cat3(a2, 1536, U + O_ekq, 512, 512, U + O_skq, 512, 1024,
                     U + O_resk, 512, U + O_Wks2, roww, ncolw, off0, koffB);
      v8f a1 = {};
      a1 = gemm_cat3(a1, 1024, U + O_skq, 512, 512, h_lds, 512, 1024,
                     h_lds, 512, U + O_Wks1, roww, ncolw, off0, koffB);
      float b2v = bks2[ncolw], b1v = bks1[ncolw];
#pragma unroll
      for (int r = 0; r < 8; ++r)
        U[O_ks + (size_t)(rbasew + r) * 512 + ncolw] =
            f2bf(sigf(a2[r] + b2v) * tanhf(a1[r] + b1v));
    } else {
      v8f a = {};
      a = gemm_cat3(a, 1536, U + O_resk, 512, 512, U + O_skq, 512, 1024,
                    U + O_ekq, 512, U + O_Wki1, roww, ncolw, off0, koffB);
      float bv = bki1[ncolw];
#pragma unroll
      for (int r = 0; r < 8; ++r)
        U[O_u + (size_t)(rbasew + r) * 512 + ncolw] =
            f2bf((tanhf(a[r] + bv) + 1.0f) * 0.5f);
    }
    target += 16; gbar(cnt, target);

    // ---- stage C: ka = sig([ekq|ks]@Wka2+b) * tanh(resk) ----
    if (wave < 8) {
      v8f a = {};
      a = gemm_cat3(a, 1024, U + O_ekq, 512, 512, U + O_ks, 512, 1024,
                    U + O_ks, 512, U + O_Wka2, roww, ncolw, off0, koffB);
      float bv = bka2[ncolw];
#pragma unroll
      for (int r = 0; r < 8; ++r) {
        int rr = rbasew + r;
        float th = tanhf(bf2f(U[O_resk + (size_t)rr * 512 + ncolw]));
        U[O_ka + (size_t)rr * 512 + ncolw] = f2bf(sigf(a[r] + bv) * th);
      }
    }
    target += 16; gbar(cnt, target);

    // ---- stage D: ki & h_pre (waves 0-7), mlp1 (WGs 0-3, waves 8-15) ----
    if (t < 199 && wave < 8) {
      const u16t* qab = U + O_qa + (size_t)t * 256;
      v8f a = {};
      a = gemm_cat3(a, 1280, U + O_ks, 512, 512, U + O_ka, 512, 1024,
                    qab, QCS, U + O_Wki2, roww, ncolw, off0, koffB);
      float bv = bki2[ncolw];
#pragma unroll
      for (int r = 0; r < 8; ++r) {
        int rr = rbasew + r;
        float ki = sigf(a[r] + bv) * bf2f(U[O_u + (size_t)rr * 512 + ncolw]);
        float hp = F[F_hf + (size_t)rr * 512 + ncolw] + ki;
        F[F_hf + (size_t)rr * 512 + ncolw] = hp;
        U[O_hbf + (size_t)rr * 512 + ncolw] = f2bf(hp);   // h_pre for fo's A input
      }
    }
    if (t >= 1 && wg < 4 && wave >= 8) {
      const int ts3 = wave - 8;
      const int m0m = (ts3 >> 1) * 16;
      const int rowm = m0m + (lane & 15);
      const int ncolm = wg * 32 + (ts3 & 1) * 16 + (lane & 15);
      const int rbasem = m0m + ((lane < 16) ? 0 : 8);
      v8f a = {};
      a = gemm_cat3(a, 768, U + O_ka, 512, 512, qcb, QCS, 768,
                    qcb, QCS, U + O_Wm1, rowm, ncolm, off0, koffB);
      float bv = bm1[ncolm];
#pragma unroll
      for (int r = 0; r < 8; ++r) {
        float m = a[r] + bv;
        U[O_m1a + (size_t)(rbasem + r) * 128 + ncolm] = f2bf(m > 0.f ? m : 0.f);
      }
    }
    target += 16; gbar(cnt, target);

    // ---- stage E: h = h_pre*sig(fo) (waves 0-7); prediction (waves 8-11) ----
    if (t < 199 && wave < 8) {
      const u16t* qnb = U + O_qc + (size_t)(t + 1) * 256;
      const u16t* tdb = U + O_tde + (size_t)t * 256;
      v8f a = {};
      a = gemm_cat3(a, 1024, qnb, QCS, 256, tdb, TDS, 512,
                    U + O_hbf, 512, U + O_Wfo, roww, ncolw, off0, koffB);
      float bv = bfo[ncolw];
#pragma unroll
      for (int r = 0; r < 8; ++r) {
        int rr = rbasew + r;
        float hn = F[F_hf + (size_t)rr * 512 + ncolw] * sigf(a[r] + bv);
        F[F_hf + (size_t)rr * 512 + ncolw] = hn;
        U[O_hbf + (size_t)rr * 512 + ncolw] = f2bf(hn);
      }
    }
    if (t >= 1 && wave >= 8 && wave < 12) {
      const int b = wg + 16 * (wave - 8);      // one batch row per wave
      float aG = 0.f, aS = 0.f, aK = 0.f;
      for (int j = lane; j < 768; j += 32) {
        float ni = (j < 512)
            ? bf2f(U[O_ka + (size_t)b * 512 + j])
            : bf2f(U[O_qc + (size_t)b * QCS + (size_t)t * 256 + (j - 512)]);
        aG += ni * Gw[j];
        aS += ni * Sw[j];
      }
      const int kc = KCp[b * 200 + t];
      for (int j = lane; j < 128; j += 32)
        aK += bf2f(U[O_m1a + (size_t)b * 128 + j]) * w2[(size_t)j * 512 + kc];
      for (int o = 16; o > 0; o >>= 1) {
        aG += __shfl_down(aG, o, 32);
        aS += __shfl_down(aS, o, 32);
        aK += __shfl_down(aK, o, 32);
      }
      if (lane == 0) {
        float kn = sigf(aK + b2p[kc]);
        float Gd = sigf(aG + Gb[0]) * 0.5f;
        float Sd = sigf(aS + Sb[0]) * 0.5f;
        float al = F[F_alpha + b * 200 + t];
        float df = F[F_diff + b * 200 + t];
        float p  = sigf(6.808f * al * (kn - df));
        out[b * 199 + (t - 1)] = Gd * (1.0f - p) + (1.0f - Sd) * p;
      }
    }
    target += 16; gbar(cnt, target);
  }
}

// ---------------- host launcher ----------------
extern "C" void kernel_launch(void* const* d_in, const int* in_sizes, int n_in,
                              void* d_out, int out_size, void* d_ws, size_t ws_size,
                              hipStream_t stream) {
  const int*   Q       = (const int*)d_in[0];
  const int*   KC      = (const int*)d_in[1];
  const int*   R       = (const int*)d_in[2];
  const int*   TS      = (const int*)d_in[3];
  const float* Eq      = (const float*)d_in[4];
  const float* Ec      = (const float*)d_in[5];
  const float* Er      = (const float*)d_in[6];
  const float* Etd     = (const float*)d_in[7];
  const float* qdiff_w = (const float*)d_in[8];
  const float* qdiff_b = (const float*)d_in[9];
  const float* qalpha_w= (const float*)d_in[10];
  const float* qalpha_b= (const float*)d_in[11];
  const float* kinit   = (const float*)d_in[12];
  const float* skq_w   = (const float*)d_in[13];
  const float* skq_b   = (const float*)d_in[14];
  const float* ekq_w   = (const float*)d_in[15];
  const float* ekq_b   = (const float*)d_in[16];
  const float* ks1_w   = (const float*)d_in[17];
  const float* ks1_b   = (const float*)d_in[18];
  const float* ks2_w   = (const float*)d_in[19];
  const float* ks2_b   = (const float*)d_in[20];
  const float* ka2_w   = (const float*)d_in[21];
  const float* ka2_b   = (const float*)d_in[22];
  const float* ki1_w   = (const float*)d_in[23];
  const float* ki1_b   = (const float*)d_in[24];
  const float* ki2_w   = (const float*)d_in[25];
  const float* ki2_b   = (const float*)d_in[26];
  const float* fo_w    = (const float*)d_in[27];
  const float* fo_b    = (const float*)d_in[28];
  const float* mlp1_w  = (const float*)d_in[29];
  const float* mlp1_b  = (const float*)d_in[30];
  const float* mlp2_w  = (const float*)d_in[31];
  const float* mlp2_b  = (const float*)d_in[32];
  const float* pa      = (const float*)d_in[33];
  const float* pb      = (const float*)d_in[34];
  const float* G_w     = (const float*)d_in[35];
  const float* G_b     = (const float*)d_in[36];
  const float* S_w     = (const float*)d_in[37];
  const float* S_b     = (const float*)d_in[38];
  (void)in_sizes; (void)n_in; (void)out_size; (void)ws_size;

  u16t*  U   = (u16t*)d_ws;
  float* F   = (float*)((char*)d_ws + FBYTES);
  int*   cnt = (int*)(F + F_TOT);
  float* out = (float*)d_out;

  kt_zero_cnt<<<1, 32, 0, stream>>>(cnt);

  struct WJob { const float* src; size_t off; int din, dout; };
  const WJob jobs[9] = {
      {skq_w, O_Wskq, 768, 512},  {ekq_w, O_Wekq, 768, 512},
      {ks1_w, O_Wks1, 1024, 512}, {ks2_w, O_Wks2, 1536, 512},
      {ka2_w, O_Wka2, 1024, 512}, {ki1_w, O_Wki1, 1536, 512},
      {ki2_w, O_Wki2, 1280, 512}, {fo_w, O_Wfo, 1024, 512},
      {mlp1_w, O_Wm1, 768, 128}};
  for (int j = 0; j < 9; ++j) {
    int n = jobs[j].din * jobs[j].dout;
    kt_wprep<<<(n + 255) / 256, 256, 0, stream>>>(jobs[j].src, U + jobs[j].off,
                                                  jobs[j].din, jobs[j].dout);
  }

  kt_prep_acts<<<dim3(200, 64), 256, 0, stream>>>(
      Q, KC, R, TS, Eq, Ec, Er, Etd, qdiff_w, qdiff_b, qalpha_w, qalpha_b,
      pa, pb, U + O_qc, U + O_qa, U + O_tde, F + F_diff, F + F_alpha);

  kt_recur<<<16, 512, 64 * 512 * sizeof(u16t), stream>>>(
      U, F, cnt, kinit, skq_b, ekq_b, ks1_b, ks2_b, ka2_b, ki1_b, ki2_b, fo_b,
      mlp1_b, mlp2_w, mlp2_b, G_w, G_b, S_w, S_b, KC, out);
}
